// FMoE_29789893165071
// MI455X (gfx1250) — matrix operand order repaired and checked by reading, not compiled, not run on hardware
//
#include <hip/hip_runtime.h>

// ---------------------------------------------------------------------------
// MoE forward for MI455X (gfx1250, wave32, WMMA + Tensor Data Mover).
// Routed top-2 expert compute (275 GFLOP vs 1.1 TFLOP dense).
// Prep pass converts/transposes weights to bf16 (fits 192MB L2); expert kernel
// streams weight tiles with TENSOR_LOAD_TO_LDS (double buffered, TENSORcnt),
// computes with v_wmma_f32_16x16x32_bf16, accumulates fp32 in VGPRs.
// ---------------------------------------------------------------------------

typedef __bf16 bf16;
typedef __attribute__((ext_vector_type(16))) __bf16 v16bf;
typedef __attribute__((ext_vector_type(8)))  float  v8f;
typedef __attribute__((ext_vector_type(4)))  unsigned int v4u;
typedef __attribute__((ext_vector_type(8)))  int v8i;
typedef __attribute__((ext_vector_type(4)))  int v4i;

constexpr int T = 8192, D = 1024, E = 8, DFF = 4096;
constexpr int MT = 32;                 // routed rows per expert block
constexpr int XS_STRIDE = D + 8;       // bf16 elems, pad vs bank conflicts
constexpr int HS_STRIDE = 128 + 8;
constexpr int WS_K      = 128 + 8;     // 136: produced by TDM pad (16B / 256B row)
constexpr int WS_BYTES  = 128 * WS_K * 2;   // one 128x128 bf16 tile, padded
constexpr int SMEM_BYTES = MT*XS_STRIDE*2 + MT*HS_STRIDE*2 + 2*WS_BYTES + MT*4;

union Frag { v16bf v; unsigned int u[8]; };

__device__ __forceinline__ float gelu_tanh(float v) {
  float c = 0.7978845608028654f * (v + 0.044715f * v * v * v);
  return 0.5f * v * (1.0f + tanhf(c));
}

// ---------------------------------------------------------------------------
// TDM: load one 128x128 bf16 tile (row stride = row_stride elems in global)
// into LDS at lds_off, padded +16B per 256B row -> LDS stride 136 bf16.
// D# per CDNA5 ISA; groups 2/3 zero (2D tile).
// ---------------------------------------------------------------------------
__device__ __forceinline__ void tdm_load_tile(unsigned lds_off, const bf16* gsrc,
                                              unsigned long long row_stride) {
  unsigned long long ga = (unsigned long long)(uintptr_t)gsrc;
  v4u g0;
  g0[0] = 1u;                                     // count=1 (valid user D#)
  g0[1] = lds_off;                                // lds_addr (bytes)
  g0[2] = (unsigned)(ga & 0xFFFFFFFFu);           // global_addr[31:0]
  g0[3] = (unsigned)((ga >> 32) & 0x01FFFFFFu)    // global_addr[56:32]
        | (2u << 30);                             // type=2 ("image")
  v8i g1;
  const unsigned flags = (1u << 16)               // data_size=1 -> 2 bytes
                       | (1u << 20)               // pad_enable
                       | (5u << 22)               // pad_interval: 64 dwords (256B)
                       | (3u << 25);              // pad_amount: 4 dwords (16B)
  const unsigned td0 = 1u << 20, td1 = 1u << 20;  // tensor dims >> tile: no OOB
  g1[0] = (int)flags;
  g1[1] = (int)((td0 & 0xFFFFu) << 16);           // abar_addr=0 | tensor_dim0 lo
  g1[2] = (int)((td0 >> 16) | ((td1 & 0xFFFFu) << 16));
  g1[3] = (int)((td1 >> 16) | (128u << 16));      // tile_dim0 = 128
  g1[4] = (int)128u;                              // tile_dim1 = 128, tile_dim2 = 0
  g1[5] = (int)(unsigned)(row_stride & 0xFFFFFFFFull);       // dim0_stride lo
  g1[6] = (int)(unsigned)((row_stride >> 32) & 0xFFFFull);   // dim0_stride hi|s1 lo=0
  g1[7] = 0;
  v4i gz4 = {0, 0, 0, 0};
  v8i gz8 = {0, 0, 0, 0, 0, 0, 0, 0};
  // 6-arg form (this toolchain's signature): (g0, g1, g2, g3, g?, cpol)
  __builtin_amdgcn_tensor_load_to_lds(g0, g1, gz4, gz4, gz8, 0);
}

// ---------------------------------------------------------------------------
// Prep: fp32 [R][C] -> bf16 [C][R] transpose via 32x32 LDS tile. block (32,8).
// ---------------------------------------------------------------------------
__global__ void moe_transpose_cvt(const float* __restrict__ src, bf16* __restrict__ dst,
                                  int R, int C) {
  __shared__ float tile[32][33];
  const int e = blockIdx.z;
  const float* s = src + (size_t)e * R * C;
  bf16* d = dst + (size_t)e * R * C;
  const int cx = blockIdx.x * 32, ry = blockIdx.y * 32;
  const int tx = threadIdx.x, ty = threadIdx.y;
#pragma unroll
  for (int i = 0; i < 4; ++i)
    tile[ty + 8*i][tx] = s[(size_t)(ry + ty + 8*i) * C + cx + tx];
  __syncthreads();
#pragma unroll
  for (int i = 0; i < 4; ++i)
    d[(size_t)(cx + ty + 8*i) * R + ry + tx] = (bf16)tile[tx][ty + 8*i];
}

// ---------------------------------------------------------------------------
// Routing counters init
// ---------------------------------------------------------------------------
__global__ void moe_init(int* counts, int* cursor) {
  if (threadIdx.x < E) { counts[threadIdx.x] = 0; cursor[threadIdx.x] = 0; }
}

// ---------------------------------------------------------------------------
// Gate: one wave32 per token, 8 dots of length 1024, shfl_xor reduce, top-2.
// ---------------------------------------------------------------------------
__global__ void moe_gate(const float* __restrict__ x, const float* __restrict__ Wg,
                         const float* __restrict__ bg, int* __restrict__ top_idx,
                         float* __restrict__ top_score, int* __restrict__ counts) {
  const int wave = threadIdx.x >> 5, lane = threadIdx.x & 31;
  const int t = blockIdx.x * 8 + wave;
  if (t >= T) return;
  float p[E];
#pragma unroll
  for (int e = 0; e < E; ++e) p[e] = 0.f;
  const float* xr = x + (size_t)t * D;
#pragma unroll 4
  for (int i = 0; i < D / 32; ++i) {
    int d = i * 32 + lane;
    float xv = xr[d];
    const float4* wq = (const float4*)(Wg + (size_t)d * E);
    float4 wa = wq[0], wb = wq[1];
    p[0] += xv * wa.x; p[1] += xv * wa.y; p[2] += xv * wa.z; p[3] += xv * wa.w;
    p[4] += xv * wb.x; p[5] += xv * wb.y; p[6] += xv * wb.z; p[7] += xv * wb.w;
  }
#pragma unroll
  for (int off = 16; off >= 1; off >>= 1) {
#pragma unroll
    for (int e = 0; e < E; ++e) p[e] += __shfl_xor(p[e], off);
  }
  if (lane == 0) {
    float lg[E];
#pragma unroll
    for (int e = 0; e < E; ++e) lg[e] = p[e] + bg[e];
    int i0 = 0; float v0 = lg[0];
#pragma unroll
    for (int e = 1; e < E; ++e) if (lg[e] > v0) { v0 = lg[e]; i0 = e; }
    int i1 = -1; float v1 = -3.4e38f;
#pragma unroll
    for (int e = 0; e < E; ++e) if (e != i0 && lg[e] > v1) { v1 = lg[e]; i1 = e; }
    float s1 = __expf(v1 - v0);
    float inv = 1.f / (1.f + s1);
    top_idx[2*t] = i0;  top_idx[2*t+1] = i1;
    top_score[2*t] = inv; top_score[2*t+1] = s1 * inv;
    atomicAdd(&counts[i0], 1);
    atomicAdd(&counts[i1], 1);
  }
}

__global__ void moe_prefix(const int* __restrict__ counts, int* __restrict__ offsets,
                           int* __restrict__ cursor) {
  if (threadIdx.x == 0) {
    int acc = 0;
    for (int e = 0; e < E; ++e) { offsets[e] = acc; acc += counts[e]; }
    offsets[E] = acc;
  }
  if (threadIdx.x < E) cursor[threadIdx.x] = 0;
}

__global__ void moe_scatter(const int* __restrict__ top_idx, const int* __restrict__ offsets,
                            int* __restrict__ cursor, int* __restrict__ tok_of_slot,
                            int* __restrict__ slot_of_tok) {
  int t = blockIdx.x * blockDim.x + threadIdx.x;
  if (t >= T) return;
#pragma unroll
  for (int j = 0; j < 2; ++j) {
    int e = top_idx[2*t + j];
    int slot = offsets[e] + atomicAdd(&cursor[e], 1);
    tok_of_slot[slot] = t;
    slot_of_tok[2*t + j] = slot;
  }
}

// ---------------------------------------------------------------------------
// Fused expert MLP over a 32-row tile of one expert's segment.
// Weights arrive via TDM (wave 0 issues, double-buffered WsA/WsB, TENSORcnt).
// ---------------------------------------------------------------------------
__global__ void __launch_bounds__(256) moe_expert(
    const float* __restrict__ x,   const bf16* __restrict__ W1T,
    const bf16* __restrict__ W2T,  const float* __restrict__ b1,
    const float* __restrict__ b2,  const int* __restrict__ offsets,
    const int* __restrict__ tok_of_slot, float* __restrict__ ybuf) {
  extern __shared__ char smem[];
  bf16* Xs  = (bf16*)smem;                                          // [32][1032]
  bf16* Hs  = (bf16*)(smem + MT*XS_STRIDE*2);                       // [32][136]
  bf16* WsA = (bf16*)(smem + MT*XS_STRIDE*2 + MT*HS_STRIDE*2);      // [128][136]
  bf16* WsB = (bf16*)(smem + MT*XS_STRIDE*2 + MT*HS_STRIDE*2 + WS_BYTES);
  int* tokids = (int*)(smem + MT*XS_STRIDE*2 + MT*HS_STRIDE*2 + 2*WS_BYTES);

  const int e   = blockIdx.y;
  const int mt  = blockIdx.x;
  const int seg = offsets[e];
  const int cnt = offsets[e+1] - seg;
  if (mt * MT >= cnt) return;                   // block-uniform
  const int rows = min(MT, cnt - mt * MT);

  const int tid  = threadIdx.x;
  const int wv   = tid >> 5;
  const int lane = tid & 31;
  const int half = lane >> 4;
  const int cl   = lane & 15;

  const bf16* W1Te = W1T + (size_t)e * DFF * D;    // [DFF][D]  (n=f rows, k=d cols)
  const bf16* W2Te = W2T + (size_t)e * D * DFF;    // [D][DFF]  (n=d rows, k=f cols)
  const float* b1e = b1 + (size_t)e * DFF;
  const float* b2e = b2 + (size_t)e * D;

  const unsigned ldsA = (unsigned)(uintptr_t)(void*)WsA;
  const unsigned ldsB = (unsigned)(uintptr_t)(void*)WsB;

  if (tid < MT) tokids[tid] = tok_of_slot[seg + mt*MT + min(tid, rows - 1)];
  __syncthreads();
  for (int i = tid; i < MT * D; i += 256) {     // gather rows + fp32->bf16
    int lr = i >> 10, col = i & (D - 1);
    Xs[lr * XS_STRIDE + col] = (bf16)x[(size_t)tokids[lr] * D + col];
  }
  __syncthreads();

  v8f acc[8][2];
#pragma unroll
  for (int nb = 0; nb < 8; ++nb)
#pragma unroll
    for (int m = 0; m < 2; ++m)
#pragma unroll
      for (int r = 0; r < 8; ++r) acc[nb][m][r] = 0.f;

  for (int c = 0; c < DFF / 128; ++c) {
    v8f hacc[2];
#pragma unroll
    for (int m = 0; m < 2; ++m)
#pragma unroll
      for (int r = 0; r < 8; ++r) hacc[m][r] = 0.f;

    // ---- GEMM1: H[32,128] = X @ W1[:, c*128..]; W1T tile rows f, cols d ----
    if (wv == 0)
      tdm_load_tile(ldsA, W1Te + (size_t)(c*128) * D, D);          // kb=0
    for (int kb = 0; kb < 8; ++kb) {
      if (wv == 0) {
        if (kb < 7) {
          tdm_load_tile((kb & 1) ? ldsA : ldsB,
                        W1Te + (size_t)(c*128) * D + (kb+1)*128, D);
          __builtin_amdgcn_s_wait_tensorcnt(1);   // current tile landed
        } else {
          __builtin_amdgcn_s_wait_tensorcnt(0);
        }
      }
      __syncthreads();
      const bf16* Wb = (kb & 1) ? WsB : WsA;
#pragma unroll
      for (int m = 0; m < 2; ++m) {
#pragma unroll
        for (int kk = 0; kk < 4; ++kk) {
          Frag A, B;
          const int arow  = 16*m + cl;                 // A: row = lane%16
          const int kbase = kb*128 + kk*32;
#pragma unroll
          for (int j = 0; j < 8; ++j) {                // A: Vj -> K=16(j/4)+8h+2(j%4)
            int koff = 16*(j >> 2) + 8*half + 2*(j & 3);
            A.u[j] = *(const unsigned int*)&Xs[arow * XS_STRIDE + kbase + koff];
          }
          const int bcol = wv*16 + cl;                 // B: col=lane%16, K=16(j/4)+8h+2(j%4)
          const int bk   = kk*32 + 8*half;
#pragma unroll
          for (int j = 0; j < 8; ++j)
            B.u[j] = *(const unsigned int*)&Wb[bcol * WS_K + bk + 16*(j >> 2) + 2*(j & 3)];
          hacc[m] = __builtin_amdgcn_wmma_f32_16x16x32_bf16(
              false, A.v, false, B.v, (short)0, hacc[m], false, false);
        }
      }
      __syncthreads();
    }

    // prefetch GEMM2 nb=0 tile while we do the GELU epilogue
    if (wv == 0)
      tdm_load_tile(ldsA, W2Te + (size_t)0 * DFF + c*128, DFF);

    // ---- bias + GELU -> Hs (bf16). C/D layout: Vr -> row = r + 8*half ----
    {
      float bv = b1e[c*128 + wv*16 + cl];
#pragma unroll
      for (int m = 0; m < 2; ++m)
#pragma unroll
        for (int r = 0; r < 8; ++r) {
          int row = 16*m + r + 8*half;
          Hs[row * HS_STRIDE + wv*16 + cl] = (bf16)gelu_tanh(hacc[m][r] + bv);
        }
    }
    __syncthreads();

    // ---- GEMM2: Y[32,1024] += H @ W2[c*128.., :]; W2T tile rows d, cols f --
#pragma unroll
    for (int nb = 0; nb < 8; ++nb) {
      if (wv == 0) {
        if (nb < 7) {
          tdm_load_tile((nb & 1) ? ldsA : ldsB,
                        W2Te + (size_t)((nb+1)*128) * DFF + c*128, DFF);
          __builtin_amdgcn_s_wait_tensorcnt(1);
        } else {
          __builtin_amdgcn_s_wait_tensorcnt(0);
        }
      }
      __syncthreads();
      const bf16* Wb = (nb & 1) ? WsB : WsA;
#pragma unroll
      for (int m = 0; m < 2; ++m) {
#pragma unroll
        for (int kk = 0; kk < 4; ++kk) {
          Frag A, B;
          const int arow = 16*m + cl;
#pragma unroll
          for (int j = 0; j < 8; ++j) {
            int koff = 16*(j >> 2) + 8*half + 2*(j & 3);
            A.u[j] = *(const unsigned int*)&Hs[arow * HS_STRIDE + kk*32 + koff];
          }
          const int bcol = wv*16 + cl;
          const int bk   = kk*32 + 8*half;
#pragma unroll
          for (int j = 0; j < 8; ++j)
            B.u[j] = *(const unsigned int*)&Wb[bcol * WS_K + bk + 16*(j >> 2) + 2*(j & 3)];
          acc[nb][m] = __builtin_amdgcn_wmma_f32_16x16x32_bf16(
              false, A.v, false, B.v, (short)0, acc[nb][m], false, false);
        }
      }
      __syncthreads();
    }
  }

  // ---- epilogue: y + b2 -> ybuf[slot] ----
#pragma unroll
  for (int nb = 0; nb < 8; ++nb) {
    const int col = (nb*8 + wv) * 16 + cl;
    const float b2v = b2e[col];
#pragma unroll
    for (int m = 0; m < 2; ++m)
#pragma unroll
      for (int r = 0; r < 8; ++r) {
        int lr = 16*m + r + 8*half;
        if (lr < rows)
          ybuf[(size_t)(seg + mt*MT + lr) * D + col] = acc[nb][m][r] + b2v;
      }
  }
}

// ---------------------------------------------------------------------------
// Combine: out[t] = s0*y[slot0] + s1*y[slot1]   (float4, deterministic)
// ---------------------------------------------------------------------------
__global__ void moe_combine(const float* __restrict__ ybuf, const int* __restrict__ slot_of_tok,
                            const float* __restrict__ top_score, float* __restrict__ out) {
  int idx = blockIdx.x * blockDim.x + threadIdx.x;
  if (idx >= T * (D / 4)) return;
  int t = idx >> 8, c = idx & 255;
  const float4* yb = (const float4*)ybuf;
  int   s0 = slot_of_tok[2*t], s1 = slot_of_tok[2*t + 1];
  float w0 = top_score[2*t],  w1 = top_score[2*t + 1];
  float4 a = yb[(size_t)s0 * 256 + c];
  float4 b = yb[(size_t)s1 * 256 + c];
  float4 o;
  o.x = w0*a.x + w1*b.x; o.y = w0*a.y + w1*b.y;
  o.z = w0*a.z + w1*b.z; o.w = w0*a.w + w1*b.w;
  ((float4*)out)[(size_t)t * 256 + c] = o;
}

// ---------------------------------------------------------------------------
// Host launcher. Workspace layout:
//   [0]      top_idx   int[2T]  64KB     [64K]  top_score f32[2T] 64KB
//   [128K]   counts/offsets/cursor 256B
//   [128K+256]  tok_of_slot int[2T] 64KB   [+64K] slot_of_tok int[2T] 64KB
//   [512K]   W1T bf16[E*DFF*D] 64MB
//   [512K+64M]  W2T bf16[E*D*DFF] 64MB
//   [512K+128M] ybuf f32[2T*D] 64MB          total ~192.5MB
// ---------------------------------------------------------------------------
extern "C" void kernel_launch(void* const* d_in, const int* in_sizes, int n_in,
                              void* d_out, int out_size, void* d_ws, size_t ws_size,
                              hipStream_t stream) {
  const float* x  = (const float*)d_in[0];
  const float* Wg = (const float*)d_in[1];
  const float* bg = (const float*)d_in[2];
  const float* W1 = (const float*)d_in[3];
  const float* b1 = (const float*)d_in[4];
  const float* W2 = (const float*)d_in[5];
  const float* b2 = (const float*)d_in[6];
  float* out = (float*)d_out;

  char* ws = (char*)d_ws;
  int*   top_idx     = (int*)ws;
  float* top_score   = (float*)(ws + 65536);
  int*   counts      = (int*)(ws + 131072);
  int*   offsets     = (int*)(ws + 131072 + 64);
  int*   cursor      = (int*)(ws + 131072 + 128);
  int*   tok_of_slot = (int*)(ws + 131072 + 256);
  int*   slot_of_tok = (int*)(ws + 131072 + 256 + 65536);
  const size_t MB64 = (size_t)E * D * DFF * 2;      // 64 MiB
  bf16*  W1T = (bf16*)(ws + 524288);
  bf16*  W2T = (bf16*)(ws + 524288 + MB64);
  float* ybuf = (float*)(ws + 524288 + 2 * MB64);

  // prep: W1 [E][D][DFF] -> W1T [E][DFF][D] bf16 ; W2 [E][DFF][D] -> W2T [E][D][DFF]
  moe_transpose_cvt<<<dim3(DFF/32, D/32, E), dim3(32, 8), 0, stream>>>(W1, W1T, D, DFF);
  moe_transpose_cvt<<<dim3(D/32, DFF/32, E), dim3(32, 8), 0, stream>>>(W2, W2T, DFF, D);

  moe_init   <<<1, 32, 0, stream>>>(counts, cursor);
  moe_gate   <<<T / 8, 256, 0, stream>>>(x, Wg, bg, top_idx, top_score, counts);
  moe_prefix <<<1, 32, 0, stream>>>(counts, offsets, cursor);
  moe_scatter<<<(T + 255) / 256, 256, 0, stream>>>(top_idx, offsets, cursor,
                                                   tok_of_slot, slot_of_tok);
  dim3 egrid((2 * T) / MT, E);                      // worst-case tiles per expert
  moe_expert <<<egrid, 256, SMEM_BYTES, stream>>>(x, W1T, W2T, b1, b2, offsets,
                                                  tok_of_slot, ybuf);
  moe_combine<<<(T * (D / 4)) / 256, 256, 0, stream>>>(ybuf, slot_of_tok, top_score, out);
}
